// LDAformer_3728031613537
// MI455X (gfx1250) — matrix operand order repaired and checked by reading, not compiled, not run on hardware
//
// LDAformer forward for MI455X (gfx1250, wave32, WMMA).
// v_wmma_f32_16x16x32_bf16 everywhere GEMM-shaped; weights pre-packed into
// B-fragment order (2x b128 per fragment); compile-time GEMM shapes keep all
// fragments in fixed registers. bf16 outputs fused into producer epilogues so
// only the initial input needs a conversion pass.
// Agent attention simplified: softmax over singleton agent axis == 1 =>
// agent branch = v + broadcast(agent_v); na_bias is mathematically unused.
#include <hip/hip_runtime.h>

typedef __attribute__((ext_vector_type(16))) __bf16 bf16x16;
typedef __attribute__((ext_vector_type(8)))  float  f32x8;

#define TOK   65536   // B*L
#define LSEQ  4096
#define NB    16

__device__ __forceinline__ int lane_id() { return (int)(threadIdx.x & 31); }

__device__ __forceinline__ f32x8 zero8() {
  f32x8 z;
#pragma unroll
  for (int i = 0; i < 8; ++i) z[i] = 0.0f;
  return z;
}

__device__ __forceinline__ f32x8 wmma_bf16(bf16x16 a, bf16x16 b, f32x8 c) {
  return __builtin_amdgcn_wmma_f32_16x16x32_bf16(false, a, false, b, (short)0, c, false, false);
}

// A-fragment: 16x32 bf16 tile, row-major source.
// lane<16: row=lane, K=0..15; lane>=16: row=lane-16, K=16..31
__device__ __forceinline__ bf16x16 load_a_rowptr(const __bf16* rowp, bool valid) {
  int l = lane_id();
  bf16x16 r;
  if (valid) {
    const __bf16* p = rowp + ((l < 16) ? 0 : 16);
    *reinterpret_cast<uint4*>(&r)       = *reinterpret_cast<const uint4*>(p);
    *(reinterpret_cast<uint4*>(&r) + 1) = *reinterpret_cast<const uint4*>(p + 8);
  } else {
#pragma unroll
    for (int e = 0; e < 16; ++e) r[e] = (__bf16)0.0f;
  }
  return r;
}

__device__ __forceinline__ bf16x16 load_a_frag(const __bf16* a, int lda) {
  int l = lane_id();
  return load_a_rowptr(a + (size_t)(l & 15) * lda, true);
}

// B-fragment from pre-packed panel: fragment f, lane l -> 16 contiguous bf16.
__device__ __forceinline__ bf16x16 load_frag_packed(const __bf16* P, int f) {
  int l = lane_id();
  const __bf16* p = P + ((size_t)f * 32 + l) * 16;
  bf16x16 r;
  *reinterpret_cast<uint4*>(&r)       = *reinterpret_cast<const uint4*>(p);
  *(reinterpret_cast<uint4*>(&r) + 1) = *reinterpret_cast<const uint4*>(p + 8);
  return r;
}

// C/D 16x16 f32 epilogue: lane<16 -> N=lane, M=0..7 ; lane>=16 -> N=lane-16, M=8..15
// Optional f32 and/or bf16 destinations (same leading dim).
__device__ __forceinline__ void store_cfrag2(float* outf, __bf16* outb, int ldo,
                                             int row0, int col0, f32x8 acc,
                                             const float* bias, const float* resid,
                                             int do_relu) {
  int l = lane_id();
  int n = col0 + (l & 15);
  float bv = bias ? bias[n] : 0.0f;
  int rbase = row0 + ((l < 16) ? 0 : 8);
#pragma unroll
  for (int r = 0; r < 8; ++r) {
    int row = rbase + r;
    float v = acc[r] + bv;
    if (resid) v += resid[(size_t)row * ldo + n];
    if (do_relu) v = fmaxf(v, 0.0f);
    if (outf) outf[(size_t)row * ldo + n] = v;
    if (outb) outb[(size_t)row * ldo + n] = (__bf16)v;
  }
}

// ---------------- weight repacking (once, weights are static) ----------------

// Row-major KxN f32 -> B-fragment-order bf16 panel.
__global__ void repack_w_kernel(const float* __restrict__ W, __bf16* __restrict__ P,
                                int Kd, int N) {
  int idx = blockIdx.x * 256 + threadIdx.x;
  if (idx >= Kd * N) return;
  int k = idx / N, n0 = idx - k * N;
  int nt = n0 >> 4, n = n0 & 15;
  int ks = k >> 5, krel = k & 31;
  int hi = krel >> 4, e = krel & 15;
  int lane = (hi << 4) | n;
  int ntiles = N >> 4;
  P[((size_t)(ks * ntiles + nt) * 32 + lane) * 16 + e] = (__bf16)W[idx];
}

// Conv kernels (K*C, C, 3, 3): extract middle W column per tap t, build
// per-tap 64x192 panels (K=ci, N=ko) in fragment order.
__global__ void repack_conv_kernel(const float* __restrict__ kern, __bf16* __restrict__ P) {
  int idx = blockIdx.x * 256 + threadIdx.x;   // 3 taps * 64*192
  if (idx >= 3 * 12288) return;
  int t = idx / 12288, r = idx - t * 12288;
  int ci = r / 192, ko = r - ci * 192;
  int nt = ko >> 4, n = ko & 15;
  int ks = ci >> 5, krel = ci & 31;
  int hi = krel >> 4, e = krel & 15;
  int lane = (hi << 4) | n;
  P[(size_t)t * 12288 + ((size_t)(ks * 12 + nt) * 32 + lane) * 16 + e]
      = (__bf16)kern[(size_t)(ko * 64 + ci) * 9 + t * 3 + 1];
}

// ---------------- small utility kernels ----------------

__global__ void f2bf_kernel(const float* __restrict__ src, __bf16* __restrict__ dst, int n4) {
  int i = blockIdx.x * 256 + threadIdx.x;
  if (i < n4) {
    float4 v = reinterpret_cast<const float4*>(src)[i];
    union { __bf16 h[4]; uint2 u; } t;
    t.h[0] = (__bf16)v.x; t.h[1] = (__bf16)v.y; t.h[2] = (__bf16)v.z; t.h[3] = (__bf16)v.w;
    *reinterpret_cast<uint2*>(dst + (size_t)i * 4) = t.u;
  }
}

// mean over L for each (b, c): f32 source
__global__ void colmean_kernel(const float* __restrict__ src, float* __restrict__ out) {
  __shared__ float sm[256];
  int b = blockIdx.x, t = threadIdx.x;
  int c = t & 63, seg = t >> 6;
  const float* p = src + (size_t)b * LSEQ * 64;
  float s = 0.0f;
  for (int l = seg; l < LSEQ; l += 4) s += p[(size_t)l * 64 + c];
  sm[t] = s;
  __syncthreads();
  if (t < 64) out[b * 64 + t] = (sm[t] + sm[t + 64] + sm[t + 128] + sm[t + 192]) * (1.0f / LSEQ);
}

// mean over L for each (b, c): bf16 source (conv-chain activations)
__global__ void colmean_bf_kernel(const __bf16* __restrict__ src, float* __restrict__ out) {
  __shared__ float sm[256];
  int b = blockIdx.x, t = threadIdx.x;
  int c = t & 63, seg = t >> 6;
  const __bf16* p = src + (size_t)b * LSEQ * 64;
  float s = 0.0f;
  for (int l = seg; l < LSEQ; l += 4) s += (float)p[(size_t)l * 64 + c];
  sm[t] = s;
  __syncthreads();
  if (t < 64) out[b * 64 + t] = (sm[t] + sm[t + 64] + sm[t + 128] + sm[t + 192]) * (1.0f / LSEQ);
}

__global__ void dc_attn_kernel(const float* __restrict__ pooled,
                               const float* __restrict__ w1, const float* __restrict__ b1,
                               const float* __restrict__ w2, const float* __restrict__ b2,
                               float* __restrict__ cattn) {
  int b = blockIdx.x, t = threadIdx.x;
  __shared__ float hdn[16];
  __shared__ float lg[3];
  if (t < 16) {
    float s = b1[t];
    for (int c = 0; c < 64; ++c) s += pooled[b * 64 + c] * w1[t * 64 + c];
    hdn[t] = fmaxf(s, 0.0f);
  }
  __syncthreads();
  if (t < 3) {
    float s = b2[t];
    for (int j = 0; j < 16; ++j) s += hdn[j] * w2[t * 16 + j];
    lg[t] = s;
  }
  __syncthreads();
  if (t == 0) {
    float m = fmaxf(lg[0], fmaxf(lg[1], lg[2]));
    float e0 = __expf(lg[0] - m), e1 = __expf(lg[1] - m), e2 = __expf(lg[2] - m);
    float inv = 1.0f / (e0 + e1 + e2);
    cattn[b * 3 + 0] = e0 * inv; cattn[b * 3 + 1] = e1 * inv; cattn[b * 3 + 2] = e2 * inv;
  }
}

__global__ void add_agent_kernel(float* __restrict__ vg, const float* __restrict__ agv) {
  int i = blockIdx.x * 256 + threadIdx.x;   // TOK*64 elements
  int c = i & 63;
  int b = i >> 18;                          // / (4096*64)
  vg[i] += agv[(b << 6) + c];
}

// LayerNorm over 64 channels: out = LN(x (+ y)) * g + b ; optional bf16 copy.
__global__ void ln_kernel(const float* __restrict__ x, const float* __restrict__ y,
                          const float* __restrict__ g, const float* __restrict__ bta,
                          float* __restrict__ out, __bf16* __restrict__ outb) {
  int tok = blockIdx.x * blockDim.x + threadIdx.x;
  const float4* px = reinterpret_cast<const float4*>(x + (size_t)tok * 64);
  const float4* py = y ? reinterpret_cast<const float4*>(y + (size_t)tok * 64) : nullptr;
  float4 buf[16];
  float mean = 0.0f;
#pragma unroll
  for (int i = 0; i < 16; ++i) {
    float4 t4 = px[i];
    if (py) { float4 u = py[i]; t4.x += u.x; t4.y += u.y; t4.z += u.z; t4.w += u.w; }
    buf[i] = t4;
    mean += t4.x + t4.y + t4.z + t4.w;
  }
  mean *= (1.0f / 64.0f);
  float var = 0.0f;
#pragma unroll
  for (int i = 0; i < 16; ++i) {
    float dx = buf[i].x - mean, dy = buf[i].y - mean, dz = buf[i].z - mean, dw = buf[i].w - mean;
    var += dx * dx + dy * dy + dz * dz + dw * dw;
  }
  var *= (1.0f / 64.0f);
  float r = rsqrtf(var + 1e-5f);
  float* po = out + (size_t)tok * 64;
  __bf16* pb = outb ? outb + (size_t)tok * 64 : nullptr;
#pragma unroll
  for (int i = 0; i < 16; ++i) {
    float o0 = (buf[i].x - mean) * r * g[4 * i + 0] + bta[4 * i + 0];
    float o1 = (buf[i].y - mean) * r * g[4 * i + 1] + bta[4 * i + 1];
    float o2 = (buf[i].z - mean) * r * g[4 * i + 2] + bta[4 * i + 2];
    float o3 = (buf[i].w - mean) * r * g[4 * i + 3] + bta[4 * i + 3];
    po[4 * i + 0] = o0; po[4 * i + 1] = o1; po[4 * i + 2] = o2; po[4 * i + 3] = o3;
    if (pb) {
      union { __bf16 h[4]; uint2 u; } t;
      t.h[0] = (__bf16)o0; t.h[1] = (__bf16)o1; t.h[2] = (__bf16)o2; t.h[3] = (__bf16)o3;
      *reinterpret_cast<uint2*>(pb + 4 * i) = t.u;
    }
  }
}

__global__ void agent_attn_kernel(const float* __restrict__ h, const float* __restrict__ v,
                                  const float* __restrict__ anb, const float* __restrict__ agm,
                                  float* __restrict__ agv) {
  int bh = blockIdx.x;
  int b = bh >> 3, hh = bh & 7;
  int t = threadIdx.x;
  __shared__ float red[256];
  __shared__ float stot;
  const float scale = 0.35355339059327373f;   // 8^-0.5
  float a[8];
#pragma unroll
  for (int d = 0; d < 8; ++d) a[d] = agm[b * 64 + hh * 8 + d] * scale;
  const float* hb = h + (size_t)b * LSEQ * 64 + hh * 8;
  const float* vb = v + (size_t)b * LSEQ * 64 + hh * 8;
  const float* bb = anb + hh * LSEQ;
  float mx = -1e30f;
  for (int n = t; n < LSEQ; n += 256) {
    float s = bb[n];
#pragma unroll
    for (int d = 0; d < 8; ++d) s += a[d] * hb[(size_t)n * 64 + d];
    mx = fmaxf(mx, s);
  }
  red[t] = mx; __syncthreads();
  for (int o = 128; o > 0; o >>= 1) { if (t < o) red[t] = fmaxf(red[t], red[t + o]); __syncthreads(); }
  float gmax = red[0]; __syncthreads();
  float vac[8];
#pragma unroll
  for (int d = 0; d < 8; ++d) vac[d] = 0.0f;
  float ssum = 0.0f;
  for (int n = t; n < LSEQ; n += 256) {
    float s = bb[n];
#pragma unroll
    for (int d = 0; d < 8; ++d) s += a[d] * hb[(size_t)n * 64 + d];
    float p = __expf(s - gmax);
    ssum += p;
#pragma unroll
    for (int d = 0; d < 8; ++d) vac[d] += p * vb[(size_t)n * 64 + d];
  }
  red[t] = ssum; __syncthreads();
  for (int o = 128; o > 0; o >>= 1) { if (t < o) red[t] += red[t + o]; __syncthreads(); }
  if (t == 0) stot = red[0];
  __syncthreads();
  float inv = 1.0f / stot;
  for (int d = 0; d < 8; ++d) {
    red[t] = vac[d]; __syncthreads();
    for (int o = 128; o > 0; o >>= 1) { if (t < o) red[t] += red[t + o]; __syncthreads(); }
    if (t == 0) agv[b * 64 + hh * 8 + d] = red[0] * inv;
    __syncthreads();
  }
}

// ---------------- WMMA kernels ----------------

// out(TOK x NN) = A_bf16(TOK x KD) @ Wp(packed) + bias (+resid)(relu?).
// Dual f32/bf16 outputs (either may be null).
template <int KD, int NN>
__global__ void gemm_bias_kernel(const __bf16* __restrict__ A, const __bf16* __restrict__ Wp,
                                 const float* __restrict__ bias, const float* __restrict__ resid,
                                 float* __restrict__ outf, __bf16* __restrict__ outb,
                                 int do_relu) {
  constexpr int KST = KD / 32;
  constexpr int NTILES = NN / 16;
  int wave = threadIdx.x >> 5;
  int tok0 = blockIdx.x * 64;
  int mt = wave & 3;
  if (tok0 + 64 < TOK) __builtin_prefetch(A + (size_t)(tok0 + 64) * KD, 0, 1);
  bf16x16 afr[KST];
#pragma unroll
  for (int ks = 0; ks < KST; ++ks)
    afr[ks] = load_a_frag(A + (size_t)(tok0 + mt * 16) * KD + ks * 32, KD);
#pragma unroll
  for (int nn = 0; nn < NTILES / 2; ++nn) {
    int nt = (wave >> 2) + 2 * nn;
    f32x8 acc = zero8();
#pragma unroll
    for (int ks = 0; ks < KST; ++ks)
      acc = wmma_bf16(afr[ks], load_frag_packed(Wp, ks * NTILES + nt), acc);
    store_cfrag2(outf, outb, NN, tok0 + mt * 16, nt * 16, acc, bias, resid, do_relu);
  }
}

// Dynamic 3-tap conv (64 -> 3*64) with fused kernel-mixture epilogue; bf16 out
// (its only consumers are the next conv's A-matrix and the pooled mean).
__global__ void dyn_conv_kernel(const __bf16* __restrict__ xbf, const __bf16* __restrict__ Wp,
                                const float* __restrict__ cattn, __bf16* __restrict__ out) {
  int wave = threadIdx.x >> 5;
  int tok0 = blockIdx.x * 64;
  int b = tok0 >> 12, bL = b << 12;
  float w0 = cattn[b * 3 + 0], w1 = cattn[b * 3 + 1], w2 = cattn[b * 3 + 2];
  int mt = wave & 3, j0 = wave >> 2;
  f32x8 acc[2][3];
#pragma unroll
  for (int q = 0; q < 2; ++q)
#pragma unroll
    for (int kg = 0; kg < 3; ++kg) acc[q][kg] = zero8();

#pragma unroll
  for (int t = 0; t < 3; ++t) {
    bf16x16 afr[2];
#pragma unroll
    for (int ks = 0; ks < 2; ++ks) {
      int l = lane_id();
      int g = tok0 + mt * 16 + (l & 15) + (t - 1);
      bool valid = (g >= bL) && (g < bL + LSEQ);
      afr[ks] = load_a_rowptr(xbf + (size_t)(valid ? g : 0) * 64 + ks * 32, valid);
    }
    const __bf16* tp = Wp + t * 12288;
#pragma unroll
    for (int q = 0; q < 2; ++q) {
      int j = j0 + 2 * q;
#pragma unroll
      for (int ks = 0; ks < 2; ++ks)
#pragma unroll
        for (int kg = 0; kg < 3; ++kg)
          acc[q][kg] = wmma_bf16(afr[ks], load_frag_packed(tp, ks * 12 + kg * 4 + j),
                                 acc[q][kg]);
    }
  }
#pragma unroll
  for (int q = 0; q < 2; ++q) {
    int j = j0 + 2 * q;
    f32x8 comb;
#pragma unroll
    for (int r = 0; r < 8; ++r)
      comb[r] = w0 * acc[q][0][r] + w1 * acc[q][1][r] + w2 * acc[q][2][r];
    store_cfrag2(nullptr, out, 64, tok0 + mt * 16, j * 16, comb, nullptr, nullptr, 0);
  }
}

// Windowed local attention for one (batch, 64-token window).
// qkv projection (WMMA, packed weights) -> per-head scores (WMMA, K padded)
// -> vectorized LDS softmax -> P @ V^T (WMMA). V stored transposed in LDS so
// every LDS fragment access is contiguous b128. bf16 output (feeds wo GEMM).
__global__ void win_attn_kernel(const __bf16* __restrict__ hbf, const __bf16* __restrict__ Wp,
                                const float* __restrict__ bq, const float* __restrict__ bk,
                                const float* __restrict__ bv, __bf16* __restrict__ lo) {
  __shared__ __attribute__((aligned(16))) __bf16 smem[28672];
  __bf16* sP  = smem;           // 4 heads x 64x64 scores/probs (row-major)
  __bf16* sQ  = smem + 16384;   // 64x64 (token, dim)
  __bf16* sK  = smem + 20480;   // 64x64 (token, dim)
  __bf16* sVt = smem + 24576;   // 64x64 transposed (dim, token)
  int tid = threadIdx.x, wave = tid >> 5;   // 128 threads, 4 waves
  int tok0 = blockIdx.x * 64;
  int l = lane_id();

  // ---- q/k/v = x @ W + b ----
  {
    int mt = wave;
    bf16x16 afr[2];
#pragma unroll
    for (int ks = 0; ks < 2; ++ks)
      afr[ks] = load_a_frag(hbf + (size_t)(tok0 + mt * 16) * 64 + ks * 32, 64);
#pragma unroll
    for (int which = 0; which < 3; ++which) {
      const float* bp = (which == 0) ? bq : ((which == 1) ? bk : bv);
#pragma unroll
      for (int nt = 0; nt < 4; ++nt) {
        f32x8 acc = zero8();
#pragma unroll
        for (int ks = 0; ks < 2; ++ks)
          acc = wmma_bf16(afr[ks], load_frag_packed(Wp + which * 4096, ks * 4 + nt), acc);
        int n = nt * 16 + (l & 15);
        float bvv = bp[n];
        int rbase = mt * 16 + ((l < 16) ? 0 : 8);
        if (which == 2) {
          union { __bf16 h[8]; uint4 u; } pk;
#pragma unroll
          for (int r = 0; r < 8; ++r) pk.h[r] = (__bf16)(acc[r] + bvv);
          *reinterpret_cast<uint4*>(sVt + n * 64 + rbase) = pk.u;   // transposed, packed
        } else {
          __bf16* dp = (which == 0) ? sQ : sK;
#pragma unroll
          for (int r = 0; r < 8; ++r) dp[(rbase + r) * 64 + n] = (__bf16)(acc[r] + bvv);
        }
      }
    }
  }
  __syncthreads();

  // ---- per-head scores s = (q_h @ k_h^T) * 0.25, dk=16 padded to K=32 ----
  int hh = wave;
#pragma unroll
  for (int mt = 0; mt < 4; ++mt) {
    bf16x16 afq;
    {
      int row = mt * 16 + (l & 15);
      int kb = (l < 16) ? 0 : 8;
      *reinterpret_cast<uint4*>(&afq) =
          *reinterpret_cast<const uint4*>(sQ + row * 64 + hh * 16 + kb);
#pragma unroll
      for (int e = 8; e < 16; ++e) afq[e] = (__bf16)0.0f;
    }
#pragma unroll
    for (int jt = 0; jt < 4; ++jt) {
      bf16x16 bfk;
      {
        const __bf16* p = sK + (jt * 16 + (l & 15)) * 64 + hh * 16 + ((l < 16) ? 0 : 8);
        *reinterpret_cast<uint4*>(&bfk)       = *reinterpret_cast<const uint4*>(p);
#pragma unroll
        for (int e = 8; e < 16; ++e) bfk[e] = (__bf16)0.0f;
      }
      f32x8 acc = wmma_bf16(afq, bfk, zero8());
      int col = jt * 16 + (l & 15);
      int rbase = mt * 16 + ((l < 16) ? 0 : 8);
      __bf16* sS = sP + hh * 4096;
#pragma unroll
      for (int r = 0; r < 8; ++r) sS[(rbase + r) * 64 + col] = (__bf16)(acc[r] * 0.25f);
    }
  }
  __syncthreads();

  // ---- row softmax (b128 LDS traffic) ----
  for (int rr = tid; rr < 256; rr += 128) {
    __bf16* p = sP + (rr >> 6) * 4096 + (rr & 63) * 64;
    union { uint4 q[8]; __bf16 h[64]; } row;
#pragma unroll
    for (int j = 0; j < 8; ++j) row.q[j] = reinterpret_cast<uint4*>(p)[j];
    float m = -1e30f;
#pragma unroll
    for (int j = 0; j < 64; ++j) m = fmaxf(m, (float)row.h[j]);
    float s = 0.0f;
    float ex[64];
#pragma unroll
    for (int j = 0; j < 64; ++j) { ex[j] = __expf((float)row.h[j] - m); s += ex[j]; }
    float inv = 1.0f / s;
#pragma unroll
    for (int j = 0; j < 64; ++j) row.h[j] = (__bf16)(ex[j] * inv);
#pragma unroll
    for (int j = 0; j < 8; ++j) reinterpret_cast<uint4*>(p)[j] = row.q[j];
  }
  __syncthreads();

  // ---- out_h = P_h(64x64) @ v_h(64x16), V read transposed/contiguous ----
#pragma unroll
  for (int mt = 0; mt < 4; ++mt) {
    f32x8 acc = zero8();
#pragma unroll
    for (int ks = 0; ks < 2; ++ks) {
      bf16x16 af = load_a_frag(sP + hh * 4096 + mt * 16 * 64 + ks * 32, 64);
      bf16x16 bf;
      {
        int n = l & 15;
        int kb = (l < 16) ? 0 : 16;
        const __bf16* p = sVt + (hh * 16 + n) * 64 + ks * 32 + kb;
        *reinterpret_cast<uint4*>(&bf)       = *reinterpret_cast<const uint4*>(p);
        *(reinterpret_cast<uint4*>(&bf) + 1) = *reinterpret_cast<const uint4*>(p + 8);
      }
      acc = wmma_bf16(af, bf, acc);
    }
    store_cfrag2(nullptr, lo, 64, tok0 + mt * 16, hh * 16, acc, nullptr, nullptr, 0);
  }
}

// ---------------- host orchestration ----------------

extern "C" void kernel_launch(void* const* d_in, const int* in_sizes, int n_in,
                              void* d_out, int out_size, void* d_ws, size_t ws_size,
                              hipStream_t stream) {
  (void)in_sizes; (void)n_in; (void)out_size; (void)ws_size;
  const float* x      = (const float*)d_in[0];
  const float* dck    = (const float*)d_in[1];
  const float* dc_w1  = (const float*)d_in[2];
  const float* dc_b1  = (const float*)d_in[3];
  const float* dc_w2  = (const float*)d_in[4];
  const float* dc_b2  = (const float*)d_in[5];
  const float* emb_w  = (const float*)d_in[6];
  const float* emb_b  = (const float*)d_in[7];
  const float* getv_w = (const float*)d_in[8];
  const float* getv_b = (const float*)d_in[9];
  const float* an_bias= (const float*)d_in[10];
  /* d_in[11] = na_bias: mathematically unused (softmax over singleton axis) */
  const float* wq     = (const float*)d_in[12];
  const float* bq     = (const float*)d_in[13];
  const float* wk     = (const float*)d_in[14];
  const float* bk     = (const float*)d_in[15];
  const float* wv     = (const float*)d_in[16];
  const float* bv     = (const float*)d_in[17];
  const float* wo     = (const float*)d_in[18];
  const float* bo     = (const float*)d_in[19];
  const float* ff_w1  = (const float*)d_in[20];
  const float* ff_b1  = (const float*)d_in[21];
  const float* ff_w2  = (const float*)d_in[22];
  const float* ff_b2  = (const float*)d_in[23];
  const float* ln1_g  = (const float*)d_in[24];
  const float* ln1_b  = (const float*)d_in[25];
  const float* ln2_g  = (const float*)d_in[26];
  const float* ln2_b  = (const float*)d_in[27];

  char* ws = (char*)d_ws;
  size_t off = 0;
  auto carve = [&](size_t bytes) { char* p = ws + off; off = (off + bytes + 255) & ~(size_t)255; return p; };
  float*  h     = (float*)carve((size_t)TOK * 64 * 4);
  float*  attnS = (float*)carve((size_t)TOK * 64 * 4);
  float*  h1    = (float*)carve((size_t)TOK * 64 * 4);
  float*  vg    = (float*)carve((size_t)TOK * 64 * 4);
  __bf16* cbfA  = (__bf16*)carve((size_t)TOK * 64 * 2);
  __bf16* cbfB  = (__bf16*)carve((size_t)TOK * 64 * 2);
  __bf16* hbf   = (__bf16*)carve((size_t)TOK * 64 * 2);
  __bf16* h1bf  = (__bf16*)carve((size_t)TOK * 64 * 2);
  __bf16* lobf  = (__bf16*)carve((size_t)TOK * 64 * 2);
  __bf16* f1bf  = (__bf16*)carve((size_t)TOK * 128 * 2);
  __bf16* p_conv= (__bf16*)carve((size_t)3 * 36864 * 2);
  __bf16* p_emb = (__bf16*)carve((size_t)4096 * 2);
  __bf16* p_getv= (__bf16*)carve((size_t)3 * 4096 * 2);
  __bf16* p_qkv = (__bf16*)carve((size_t)3 * 12288 * 2);
  __bf16* p_wo  = (__bf16*)carve((size_t)3 * 4096 * 2);
  __bf16* p_ff1 = (__bf16*)carve((size_t)3 * 8192 * 2);
  __bf16* p_ff2 = (__bf16*)carve((size_t)3 * 8192 * 2);
  float*  pooled= (float*)carve(NB * 64 * 4);
  float*  cattn = (float*)carve(NB * 4 * 4);
  float*  agm   = (float*)carve(NB * 64 * 4);
  float*  agv   = (float*)carve(NB * 64 * 4);

  const int G64  = TOK / 64;                 // 1024 token-tile blocks
  const int CV64 = (TOK * 64 / 4) / 256;

  // --- pack all static weights into WMMA fragment order (bf16) ---
  for (int i = 0; i < 3; ++i) {
    repack_conv_kernel<<<144, 256, 0, stream>>>(dck + (size_t)i * 3 * 64 * 64 * 9,
                                                p_conv + (size_t)i * 36864);
    repack_w_kernel<<<16, 256, 0, stream>>>(getv_w + i * 4096, p_getv + (size_t)i * 4096, 64, 64);
    repack_w_kernel<<<16, 256, 0, stream>>>(wq + i * 4096, p_qkv + (size_t)i * 12288,        64, 64);
    repack_w_kernel<<<16, 256, 0, stream>>>(wk + i * 4096, p_qkv + (size_t)i * 12288 + 4096, 64, 64);
    repack_w_kernel<<<16, 256, 0, stream>>>(wv + i * 4096, p_qkv + (size_t)i * 12288 + 8192, 64, 64);
    repack_w_kernel<<<16, 256, 0, stream>>>(wo + i * 4096, p_wo + (size_t)i * 4096, 64, 64);
    repack_w_kernel<<<32, 256, 0, stream>>>(ff_w1 + i * 8192, p_ff1 + (size_t)i * 8192, 64, 128);
    repack_w_kernel<<<32, 256, 0, stream>>>(ff_w2 + i * 8192, p_ff2 + (size_t)i * 8192, 128, 64);
  }
  repack_w_kernel<<<16, 256, 0, stream>>>(emb_w, p_emb, 64, 64);

  // --- three dynamic multi-kernel convs (bf16 activation chain) ---
  f2bf_kernel<<<CV64, 256, 0, stream>>>(x, cbfA, TOK * 16);   // only conversion pass
  __bf16* cin = cbfA;
  __bf16* cout = cbfB;
  for (int i = 0; i < 3; ++i) {
    colmean_bf_kernel<<<NB, 256, 0, stream>>>(cin, pooled);
    dc_attn_kernel<<<NB, 32, 0, stream>>>(pooled, dc_w1 + i * 16 * 64, dc_b1 + i * 16,
                                          dc_w2 + i * 3 * 16, dc_b2 + i * 3, cattn);
    dyn_conv_kernel<<<G64, 256, 0, stream>>>(cin, p_conv + (size_t)i * 36864, cattn, cout);
    __bf16* tmp = cin; cin = cout; cout = tmp;
  }

  // --- embedding: h (f32) + hbf (bf16) in one pass ---
  gemm_bias_kernel<64, 64><<<G64, 256, 0, stream>>>(cin, p_emb, emb_b, nullptr, h, hbf, 0);

  // --- transformer layers ---
  for (int i = 0; i < 3; ++i) {
    colmean_kernel<<<NB, 256, 0, stream>>>(h, agm);
    gemm_bias_kernel<64, 64><<<G64, 256, 0, stream>>>(hbf, p_getv + (size_t)i * 4096,
                                                      getv_b + i * 64, nullptr, vg, nullptr, 0);
    agent_attn_kernel<<<NB * 8, 256, 0, stream>>>(h, vg, an_bias + (size_t)i * 8 * LSEQ, agm, agv);
    add_agent_kernel<<<TOK * 64 / 256, 256, 0, stream>>>(vg, agv);
    win_attn_kernel<<<G64, 128, 0, stream>>>(hbf, p_qkv + (size_t)i * 12288,
                                             bq + i * 64, bk + i * 64, bv + i * 64, lobf);
    gemm_bias_kernel<64, 64><<<G64, 256, 0, stream>>>(lobf, p_wo + (size_t)i * 4096,
                                                      bo + i * 64, vg, attnS, nullptr, 0);
    ln_kernel<<<TOK / 256, 256, 0, stream>>>(h, attnS, ln1_g + i * 64, ln1_b + i * 64, h1, h1bf);
    gemm_bias_kernel<64, 128><<<G64, 256, 0, stream>>>(h1bf, p_ff1 + (size_t)i * 8192,
                                                       ff_b1 + i * 128, nullptr, nullptr, f1bf, 1);
    gemm_bias_kernel<128, 64><<<G64, 256, 0, stream>>>(f1bf, p_ff2 + (size_t)i * 8192,
                                                       ff_b2 + i * 64, h1, attnS, nullptr, 0);
    ln_kernel<<<TOK / 256, 256, 0, stream>>>(attnS, nullptr, ln2_g + i * 64, ln2_b + i * 64, h, hbf);
  }

  hipMemcpyAsync(d_out, h, (size_t)TOK * 64 * 4, hipMemcpyDeviceToDevice, stream);
}
